// MultiHeadAttention_59734405153249
// MI455X (gfx1250) — matrix operand order repaired and checked by reading, not compiled, not run on hardware
//
#include <hip/hip_runtime.h>
#include <hip/hip_bf16.h>

typedef _Float16 h16;
typedef __attribute__((ext_vector_type(8)))  _Float16 v8h;
typedef __attribute__((ext_vector_type(16))) _Float16 v16h;
typedef __attribute__((ext_vector_type(8)))  float    v8f;
typedef __attribute__((ext_vector_type(4)))  int      i32x4;
typedef __attribute__((ext_vector_type(8)))  int      i32x8;
typedef __attribute__((ext_vector_type(4)))  unsigned int u32x4;

// ---------------------------------------------------------------------------
// CDNA5 feature detection (compile-only loop: guarded so fallback still builds)
// ---------------------------------------------------------------------------
#if defined(__has_builtin)
#  if __has_builtin(__builtin_amdgcn_global_load_async_to_lds_b128)
#    define HAS_ASYNC_LDS 1
#  endif
#  if __has_builtin(__builtin_amdgcn_tensor_load_to_lds)
#    define HAS_TDM 1
#  endif
#  if __has_builtin(__builtin_amdgcn_s_wait_asynccnt)
#    define HAS_WAIT_ASYNC 1
#  endif
#  if __has_builtin(__builtin_amdgcn_s_wait_tensorcnt)
#    define HAS_WAIT_TENSOR 1
#  endif
#endif
#ifndef HAS_ASYNC_LDS
#define HAS_ASYNC_LDS 0
#endif
#ifndef HAS_TDM
#define HAS_TDM 0
#endif
#ifndef HAS_WAIT_ASYNC
#define HAS_WAIT_ASYNC 0
#endif
#ifndef HAS_WAIT_TENSOR
#define HAS_WAIT_TENSOR 0
#endif

#if HAS_ASYNC_LDS
// 16B async DMA global->LDS, bypassing VGPRs (ASYNCcnt). The builtin takes
// v4i pointers in AS1 (global) / AS3 (LDS); generic-LDS pointers keep the LDS
// offset in bits [31:0], so integer casts are a safe way to switch spaces.
__device__ __forceinline__ void async_copy16(const void* g, void* l) {
  __builtin_amdgcn_global_load_async_to_lds_b128(
      (__attribute__((address_space(1))) i32x4*)(unsigned long long)g,
      (__attribute__((address_space(3))) i32x4*)(unsigned)(unsigned long long)l,
      0, 0);
}
__device__ __forceinline__ void async_wait0() {
#if HAS_WAIT_ASYNC
  __builtin_amdgcn_s_wait_asynccnt(0);
#else
  asm volatile("s_wait_asynccnt 0" ::: "memory");
#endif
}
#endif

#if HAS_TDM
__device__ __forceinline__ void tensor_wait0() {
#if HAS_WAIT_TENSOR
  __builtin_amdgcn_s_wait_tensorcnt(0);
#else
  asm volatile("s_wait_tensorcnt 0" ::: "memory");
#endif
}
// TDM 2D f16 tile load (D# per ISA 8.3/8.4). tile_w/tile_h in elements,
// tensor_stride = elements between rows. Optional LDS padding so the DMA
// lands directly in a padded-stride LDS tile.
__device__ __forceinline__ void tdm_load_2d_f16(const void* gbase, void* lbase,
                                                unsigned tile_w, unsigned tile_h,
                                                unsigned tensor_stride,
                                                unsigned pad_interval_code,
                                                unsigned pad_amount_code,
                                                bool pad_en) {
  unsigned long long ga = (unsigned long long)gbase;
  unsigned la = (unsigned)(unsigned long long)lbase;
  u32x4 g0;
  g0.x = 1u;                                   // count=1 (valid user D#)
  g0.y = la;                                   // lds_addr [63:32]
  g0.z = (unsigned)ga;                         // global_addr lo
  g0.w = (unsigned)(ga >> 32) | (2u << 30);    // global_addr hi | type=2
  i32x8 g1;
  g1[0] = (int)((1u << 16) |                   // data_size=1 (2 bytes)
                (pad_en ? (1u << 20) : 0u) |   // pad_enable
                (pad_interval_code << 22) | (pad_amount_code << 25));
  g1[1] = (int)((tile_w & 0xFFFFu) << 16);                      // tensor_dim0 lo16
  g1[2] = (int)(((tile_w >> 16) & 0xFFFFu) |                    // tensor_dim0 hi16
                ((tile_h & 0xFFFFu) << 16));                    // tensor_dim1 lo16
  g1[3] = (int)(((tile_h >> 16) & 0xFFFFu) | (tile_w << 16));   // dim1 hi | tile_dim0
  g1[4] = (int)(tile_h & 0xFFFFu);                              // tile_dim1 (dim2=0)
  g1[5] = (int)tensor_stride;                                   // tensor_dim0_stride
  g1[6] = 0;
  g1[7] = 0;
  i32x4 z4 = {0, 0, 0, 0};
  i32x8 z8 = {0, 0, 0, 0, 0, 0, 0, 0};
  // 6-arg form (clang-23 / therock signature): (g0, g1, g2, g3, g?, cpol)
  __builtin_amdgcn_tensor_load_to_lds(g0, g1, z4, z4, z8, 0);
}
#endif

// ---------------------------------------------------------------------------
// WMMA helpers (CDNA5, wave32): D = A(16x32 f16) * B(32x16 f16) + C(16x16 f32)
// ---------------------------------------------------------------------------
__device__ __forceinline__ v8f wmma_f16(v16h a, v16h b, v8f c) {
  return __builtin_amdgcn_wmma_f32_16x16x32_f16(
      false, a, false, b, (short)0, c, false, false);
}

// A fragment from row-major LDS tile (rows=M, cols=K). ld in halfs (mult of 8).
__device__ __forceinline__ v16h load_frag_a(const h16* p, int ld) {
  const int lane = threadIdx.x & 31;
  const int m  = lane & 15;
  const int hi = (lane >> 4) & 1;
  const v8h* row = (const v8h*)(p + m * ld);
  v8h a0 = row[hi];       // K = 8*hi .. +7
  v8h a1 = row[2 + hi];   // K = 16+8*hi .. +7
  return __builtin_shufflevector(a0, a1, 0,1,2,3,4,5,6,7,8,9,10,11,12,13,14,15);
}

// B fragment from *transposed* LDS tile Bt[n][k] (rows=N, cols=K).
__device__ __forceinline__ v16h load_frag_b(const h16* p, int ld) {
  const int lane = threadIdx.x & 31;
  const int n  = lane & 15;
  const int hi = (lane >> 4) & 1;
  const v8h* row = (const v8h*)(p + n * ld);
  v8h b0 = row[hi];
  v8h b1 = row[2 + hi];
  return __builtin_shufflevector(b0, b1, 0,1,2,3,4,5,6,7,8,9,10,11,12,13,14,15);
}

// ---------------------------------------------------------------------------
// Generic tiled GEMM: Out[M,N] = A[M,K] @ B[K,N] + bias[N]
// ---------------------------------------------------------------------------
template <bool A_IS_F16, bool OUT_F32>
__global__ void __launch_bounds__(256)
gemm_bias_kernel(const void* __restrict__ Aptr, const float* __restrict__ B,
                 const float* __restrict__ bias, void* __restrict__ Out,
                 int M, int N, int K) {
  __shared__ h16 sA[128 * 40];
  __shared__ h16 sBt[64 * 40];

  const int tid  = threadIdx.x;
  const int wave = tid >> 5;
  const int lane = tid & 31;
  const int bm = blockIdx.y * 128;
  const int bn = blockIdx.x * 64;

  v8f acc[4] = {};

  for (int k0 = 0; k0 < K; k0 += 32) {
    if (A_IS_F16) {
      const h16* A = (const h16*)Aptr;
#if HAS_ASYNC_LDS
      for (int i = tid; i < 512; i += 256) {
        int r = i >> 2, c = (i & 3) * 8;
        async_copy16(&A[(size_t)(bm + r) * K + k0 + c], &sA[r * 40 + c]);
      }
#else
      for (int i = tid; i < 512; i += 256) {
        int r = i >> 2, c = (i & 3) * 8;
        *(v8h*)&sA[r * 40 + c] = *(const v8h*)&A[(size_t)(bm + r) * K + k0 + c];
      }
#endif
    } else {
      const float* A = (const float*)Aptr;
      for (int i = tid; i < 128 * 32; i += 256) {
        int r = i >> 5, c = i & 31;
        sA[r * 40 + c] = (h16)A[(size_t)(bm + r) * K + k0 + c];
      }
    }
    for (int i = tid; i < 32 * 64; i += 256) {
      int kk = i >> 6, n = i & 63;
      sBt[n * 40 + kk] = (h16)B[(size_t)(k0 + kk) * N + bn + n];
    }
#if HAS_ASYNC_LDS
    if (A_IS_F16) async_wait0();
#endif
    __syncthreads();

    v16h afrag = load_frag_a(&sA[wave * 16 * 40], 40);
#pragma unroll
    for (int t = 0; t < 4; ++t) {
      v16h bfrag = load_frag_b(&sBt[t * 16 * 40], 40);
      acc[t] = wmma_f16(afrag, bfrag, acc[t]);
    }
    __syncthreads();
  }

  const int hi = (lane >> 4) & 1;
  const int n0 = lane & 15;
#pragma unroll
  for (int t = 0; t < 4; ++t) {
    const int col = bn + t * 16 + n0;
    const float bv = bias[col];
#pragma unroll
    for (int r = 0; r < 8; ++r) {
      const int row = bm + wave * 16 + r + 8 * hi;
      const float val = acc[t][r] + bv;
      if (OUT_F32) ((float*)Out)[(size_t)row * N + col] = val;
      else         ((h16*)Out)[(size_t)row * N + col]   = (h16)val;
    }
  }
}

// ---------------------------------------------------------------------------
// Attention scores + softmax. One block per (b*H+h, 32-row q-tile).
// 32x1024 fp32 score block lives in LDS (CDNA5 320KB/WGP). K tiles staged by
// the Tensor Data Mover (with D# pad -> 72-half LDS stride), Q tile by async
// LDS loads. Writes attn fp32 and 33-bucket rel weights W.
// ---------------------------------------------------------------------------
#define TQ 32
#define KT 64
__global__ void __launch_bounds__(256)
attn_scores_kernel(const h16* __restrict__ Qp, const h16* __restrict__ Kp,
                   const float* __restrict__ Ek, float* __restrict__ attn,
                   float* __restrict__ Wbkt, int T) {
  __shared__ float S[TQ * 1024];
  __shared__ h16   sQ[TQ * 72];
  __shared__ h16   sK[KT * 72];
  __shared__ float sSrel[TQ * 36];
  __shared__ float sW[TQ * 33];
  __shared__ float sRed[TQ * 8];
  __shared__ float sInv[TQ];

  const int tid  = threadIdx.x;
  const int wave = tid >> 5;
  const int lane = tid & 31;
  const int bh = blockIdx.x;
  const int b  = bh >> 4;
  const int hh = bh & 15;
  const int q0 = blockIdx.y * TQ;
  const size_t rowbase = (size_t)b * T;

  // ---- stage Q tile (32 x 64 f16) ----
#if HAS_ASYNC_LDS
  {
    int r = tid >> 3, c = (tid & 7) * 8;
    async_copy16(&Qp[(rowbase + q0 + r) * 1024 + hh * 64 + c], &sQ[r * 72 + c]);
    async_wait0();
  }
#else
  for (int i = tid; i < (TQ * 64) / 8; i += 256) {
    int r = i >> 3, c = (i & 7) * 8;
    *(v8h*)&sQ[r * 72 + c] =
        *(const v8h*)&Qp[(rowbase + q0 + r) * 1024 + hh * 64 + c];
  }
#endif
  for (int i = tid; i < TQ * 33; i += 256) sW[i] = 0.0f;
  __syncthreads();

  // Srel[r][j] = q_r . Ek[j]
  for (int i = tid; i < TQ * 33; i += 256) {
    int r = i / 33, j = i % 33;
    float a = 0.0f;
    for (int d = 0; d < 64; ++d) a += (float)sQ[r * 72 + d] * Ek[j * 64 + d];
    sSrel[r * 36 + j] = a;
  }
  __syncthreads();

  const int rb = wave & 1;
  const int cb = wave >> 1;
  for (int kt = 0; kt < T; kt += KT) {
    // ---- stage K tile (64 x 64 f16) -> padded stride 72 ----
#if HAS_TDM
    if (wave == 0) {
      // pad_interval code 4 = 32 dwords (one 64-half row);
      // pad_amount code 3 = 4 dwords (8 halfs) -> stride 72.
      tdm_load_2d_f16(&Kp[(rowbase + kt) * 1024 + hh * 64], sK,
                      /*tile_w=*/64, /*tile_h=*/KT, /*stride=*/1024,
                      /*interval=*/4u, /*amount=*/3u, /*pad=*/true);
      tensor_wait0();
    }
#else
    for (int i = tid; i < (KT * 64) / 8; i += 256) {
      int r = i >> 3, c = (i & 7) * 8;
      *(v8h*)&sK[r * 72 + c] =
          *(const v8h*)&Kp[(rowbase + kt + r) * 1024 + hh * 64 + c];
    }
#endif
    if (kt + KT < T)
      __builtin_prefetch(&Kp[(rowbase + kt + KT + (tid >> 2)) * 1024 + hh * 64], 0, 1);
    __syncthreads();

    v8f acc = {};
#pragma unroll
    for (int kc = 0; kc < 2; ++kc) {
      v16h a = load_frag_a(&sQ[rb * 16 * 72 + kc * 32], 72);
      v16h bb = load_frag_b(&sK[cb * 16 * 72 + kc * 32], 72);
      acc = wmma_f16(a, bb, acc);
    }
    const int hi = (lane >> 4) & 1;
    const int keyg = kt + cb * 16 + (lane & 15);
#pragma unroll
    for (int r = 0; r < 8; ++r) {
      const int ql = rb * 16 + r + 8 * hi;
      const int qg = q0 + ql;
      int rel = keyg - qg;
      rel = rel < -16 ? -16 : (rel > 16 ? 16 : rel);
      S[ql * 1024 + keyg] = (acc[r] + sSrel[ql * 36 + rel + 16]) * 0.125f;
    }
    __syncthreads();
  }

  // ---- softmax: 8 threads per row ----
  const int row = tid >> 3, part = tid & 7;
  float* Srow = &S[row * 1024];
  const int kbeg = part * 128, kend = kbeg + 128;
  const int qg = q0 + row;

  float pmax = -1e30f;
  for (int k = kbeg; k < kend; ++k) pmax = fmaxf(pmax, Srow[k]);
  sRed[row * 8 + part] = pmax;
  __syncthreads();
  float rowmax = sRed[row * 8];
  for (int p = 1; p < 8; ++p) rowmax = fmaxf(rowmax, sRed[row * 8 + p]);

  float ssum = 0.0f;
  for (int k = kbeg; k < kend; ++k) {
    float e = __expf(Srow[k] - rowmax);
    Srow[k] = e;
    ssum += e;
  }
  __syncthreads();
  sRed[row * 8 + part] = ssum;
  __syncthreads();
  float tot = 0.0f;
  for (int p = 0; p < 8; ++p) tot += sRed[row * 8 + p];
  const float inv = 1.0f / tot;
  if (part == 0) sInv[row] = inv;

  float* arow = attn + ((size_t)bh * T + qg) * T;
  float left = 0.0f, right = 0.0f;
  for (int k = kbeg; k < kend; ++k) {
    float p = Srow[k] * inv;
    arow[k] = p;
    if (k <= qg - 16) left += p;
    if (k >= qg + 16) right += p;
  }
  if (left  != 0.0f) atomicAdd(&sW[row * 33 + 0],  left);
  if (right != 0.0f) atomicAdd(&sW[row * 33 + 32], right);
  __syncthreads();

  for (int i = tid; i < TQ * 31; i += 256) {
    int rr = i / 31, j = (i % 31) + 1;
    int kk = (q0 + rr) - 16 + j;
    sW[rr * 33 + j] = (kk >= 0 && kk < T) ? S[rr * 1024 + kk] * sInv[rr] : 0.0f;
  }
  __syncthreads();
  for (int i = tid; i < TQ * 33; i += 256) {
    int rr = i / 33, j = i % 33;
    Wbkt[((size_t)bh * T + q0 + rr) * 33 + j] = sW[rr * 33 + j];
  }
}

// ---------------------------------------------------------------------------
// out = attn @ V (WMMA over keys) + W @ Ev (rel-value term) -> f16 concat.
// ---------------------------------------------------------------------------
__global__ void __launch_bounds__(256)
attn_out_kernel(const float* __restrict__ attn, const h16* __restrict__ Vp,
                const float* __restrict__ Ev, const float* __restrict__ Wbkt,
                h16* __restrict__ concat, int T) {
  __shared__ h16   sP[TQ * 136];
  __shared__ h16   sVt[64 * 136];
  __shared__ float sRelO[TQ * 72];
  __shared__ float sWl[TQ * 33];

  const int tid  = threadIdx.x;
  const int wave = tid >> 5;
  const int lane = tid & 31;
  const int bh = blockIdx.x;
  const int b  = bh >> 4;
  const int hh = bh & 15;
  const int q0 = blockIdx.y * TQ;

  for (int i = tid; i < TQ * 33; i += 256) {
    int r = i / 33, j = i % 33;
    sWl[r * 33 + j] = Wbkt[((size_t)bh * T + q0 + r) * 33 + j];
  }
  __syncthreads();
  for (int i = tid; i < TQ * 64; i += 256) {
    int r = i >> 6, c = i & 63;
    float a = 0.0f;
    for (int j = 0; j < 33; ++j) a += sWl[r * 33 + j] * Ev[j * 64 + c];
    sRelO[r * 72 + c] = a;
  }

  const int rb = wave & 1;
  const int cb = wave >> 1;
  v8f acc = {};

  for (int kt = 0; kt < T; kt += 128) {
    __syncthreads();
    for (int i = tid; i < TQ * 128; i += 256) {
      int r = i >> 7, k = i & 127;
      sP[r * 136 + k] = (h16)attn[((size_t)bh * T + q0 + r) * T + kt + k];
    }
    for (int i = tid; i < 128 * 64; i += 256) {
      int key = i >> 6, c = i & 63;
      sVt[c * 136 + key] = Vp[((size_t)b * T + kt + key) * 1024 + hh * 64 + c];
    }
    __syncthreads();
#pragma unroll
    for (int kc = 0; kc < 4; ++kc) {
      v16h a = load_frag_a(&sP[rb * 16 * 136 + kc * 32], 136);
      v16h bb = load_frag_b(&sVt[cb * 16 * 136 + kc * 32], 136);
      acc = wmma_f16(a, bb, acc);
    }
  }

  const int hi = (lane >> 4) & 1;
  const int c = cb * 16 + (lane & 15);
#pragma unroll
  for (int r = 0; r < 8; ++r) {
    const int ql = rb * 16 + r + 8 * hi;
    const float v = acc[r] + sRelO[ql * 72 + c];
    concat[((size_t)b * T + q0 + ql) * 1024 + hh * 64 + c] = (h16)v;
  }
}

// ---------------------------------------------------------------------------
// Launch. Inputs: 0:q 1:k 2:v 3:Wq 4:bq 5:Wk 6:bk 7:Wv 8:bv 9:Wo 10:bo 11:Ek 12:Ev
// d_out = [output (4M f32) | attn (64M f32)]
// ---------------------------------------------------------------------------
extern "C" void kernel_launch(void* const* d_in, const int* in_sizes, int n_in,
                              void* d_out, int out_size, void* d_ws, size_t ws_size,
                              hipStream_t stream) {
  const int B = 4, T = 1024, D = 1024;
  const float* q  = (const float*)d_in[0];
  const float* k  = (const float*)d_in[1];
  const float* v  = (const float*)d_in[2];
  const float* Wq = (const float*)d_in[3];
  const float* bq = (const float*)d_in[4];
  const float* Wk = (const float*)d_in[5];
  const float* bk = (const float*)d_in[6];
  const float* Wv = (const float*)d_in[7];
  const float* bv = (const float*)d_in[8];
  const float* Wo = (const float*)d_in[9];
  const float* bo = (const float*)d_in[10];
  const float* Ek = (const float*)d_in[11];
  const float* Ev = (const float*)d_in[12];

  float* out  = (float*)d_out;
  float* attn = out + (size_t)B * T * D;

  char* ws = (char*)d_ws;
  h16*   Qp     = (h16*)(ws);
  h16*   Kp     = (h16*)(ws + (size_t)8 * 1024 * 1024);
  h16*   Vp     = (h16*)(ws + (size_t)16 * 1024 * 1024);
  float* Wbkt   = (float*)(ws + (size_t)24 * 1024 * 1024);
  h16*   concat = (h16*)(ws + (size_t)33 * 1024 * 1024);

  dim3 blk(256);
  dim3 gproj(D / 64, (B * T) / 128);
  gemm_bias_kernel<false, false><<<gproj, blk, 0, stream>>>(q, Wq, bq, Qp, B * T, D, D);
  gemm_bias_kernel<false, false><<<gproj, blk, 0, stream>>>(k, Wk, bk, Kp, B * T, D, D);
  gemm_bias_kernel<false, false><<<gproj, blk, 0, stream>>>(v, Wv, bv, Vp, B * T, D, D);

  dim3 gattn(B * 16, T / TQ);
  attn_scores_kernel<<<gattn, blk, 0, stream>>>(Qp, Kp, Ek, attn, Wbkt, T);
  attn_out_kernel<<<gattn, blk, 0, stream>>>(attn, Vp, Ev, Wbkt, concat, T);

  gemm_bias_kernel<true, true><<<gproj, blk, 0, stream>>>(concat, Wo, bo, out, B * T, D, D);
}
